// MultiHeadLatentAttention_79259326480570
// MI455X (gfx1250) — matrix operand order repaired and checked by reading, not compiled, not run on hardware
//
#include <hip/hip_runtime.h>
#include <hip/hip_bf16.h>

typedef __bf16 bf16_t;
typedef __attribute__((ext_vector_type(16))) __bf16 v16bf;
typedef __attribute__((ext_vector_type(8)))  __bf16 v8bf;
typedef __attribute__((ext_vector_type(4)))  __bf16 v4bf;
typedef __attribute__((ext_vector_type(8)))  float  v8f;

#define DIM      1024
#define LATENT   512
#define HEADS    16
#define HEAD_DIM 64
#define SEQ      2048
#define BATCH    4
#define ROWS     (BATCH * SEQ)   // 8192

// ---------------------------------------------------------------------------
// CDNA5 helpers
// ---------------------------------------------------------------------------
__device__ __forceinline__ v8f wmma_bf16(v16bf a, v16bf b, v8f c) {
  return __builtin_amdgcn_wmma_f32_16x16x32_bf16(false, a, false, b,
                                                 (short)0, c, false, false);
}

// Async global->LDS copy of 16 bytes (GLOBAL_LOAD_ASYNC_TO_LDS_B128).
// LDS operand = low 32 bits of the generic shared pointer (aperture scheme:
// generic LDS addr[31:0] == LDS byte offset). Tracked by ASYNCcnt.
__device__ __forceinline__ void async_b128(bf16_t* lds_dst, const bf16_t* gsrc) {
  unsigned loff = (unsigned)(uintptr_t)(void*)lds_dst;
  unsigned long long g = (unsigned long long)(uintptr_t)gsrc;
  asm volatile("global_load_async_to_lds_b128 %0, %1, off"
               :: "v"(loff), "v"(g) : "memory");
}
__device__ __forceinline__ void wait_async_le6() {
  asm volatile("s_wait_asynccnt 0x6" ::: "memory");
}
__device__ __forceinline__ void wait_async_0() {
  asm volatile("s_wait_asynccnt 0x0" ::: "memory");
}

// A-fragment 16x32 (lane = M row; K split {0..7,16..23}/{8..15,24..31})
__device__ __forceinline__ v16bf frag_a(const bf16_t* base, int stride, int lane) {
  const bf16_t* p = base + (lane & 15) * stride + ((lane >> 4) << 3);
  v16bf f;
  v8bf* h = reinterpret_cast<v8bf*>(&f);
  h[0] = *reinterpret_cast<const v8bf*>(p);
  h[1] = *reinterpret_cast<const v8bf*>(p + 16);
  return f;
}

// B-fragment 32x16 from B^T rows (lane = N col; K split {0..7,16..23}/{8..15,24..31})
__device__ __forceinline__ v16bf frag_b(const bf16_t* base, int stride, int lane) {
  const bf16_t* p = base + (lane & 15) * stride + ((lane >> 4) << 3);
  v16bf f;
  v8bf* h = reinterpret_cast<v8bf*>(&f);
  h[0] = *reinterpret_cast<const v8bf*>(p);
  h[1] = *reinterpret_cast<const v8bf*>(p + 16);
  return f;
}

// ---------------------------------------------------------------------------
// fp32 -> bf16 elementwise (activations)
// ---------------------------------------------------------------------------
__global__ void f32_to_bf16_kernel(const float* __restrict__ in,
                                   bf16_t* __restrict__ out, int n4) {
  int i = blockIdx.x * blockDim.x + threadIdx.x;
  if (i < n4) {
    float4 v = reinterpret_cast<const float4*>(in)[i];
    v4bf o;
    o[0] = (bf16_t)v.x; o[1] = (bf16_t)v.y;
    o[2] = (bf16_t)v.z; o[3] = (bf16_t)v.w;
    reinterpret_cast<v4bf*>(out)[i] = o;
  }
}

// fp32 KxN -> bf16 NxK (weights transposed once so GEMM B-tiles are
// contiguous-K row blocks -> pure b128 LDS staging, no in-kernel transpose)
__global__ __launch_bounds__(256) void transpose_f32_to_bf16_kernel(
    const float* __restrict__ in, bf16_t* __restrict__ out, int K, int N) {
  __shared__ float tile[32][33];
  const int k0 = blockIdx.y * 32, n0 = blockIdx.x * 32;
  const int tx = threadIdx.x & 31, ty = threadIdx.x >> 5;  // 32 x 8
#pragma unroll
  for (int i = 0; i < 32; i += 8)
    tile[ty + i][tx] = in[(size_t)(k0 + ty + i) * N + n0 + tx];
  __syncthreads();
#pragma unroll
  for (int i = 0; i < 32; i += 8)
    out[(size_t)(n0 + ty + i) * K + k0 + tx] = (bf16_t)tile[tx][ty + i];
}

// ---------------------------------------------------------------------------
// bf16 GEMM: C = epilogue(A @ B), A:MxK row-major, Bt:NxK row-major (B^T).
// Block tile 128x256x32, 8 waves (2x4), wave tile 64x64 (16 WMMA / 16 LDS
// loads per wave-iteration). Double-buffered LDS staged via
// GLOBAL_LOAD_ASYNC_TO_LDS_B128 (ASYNCcnt).
// ---------------------------------------------------------------------------
__global__ __launch_bounds__(256) void gemm_bf16_kernel(
    const bf16_t* __restrict__ A, const bf16_t* __restrict__ Bt,
    void* __restrict__ C, const float* __restrict__ bias,
    int M, int N, int K, int lda, int ldb, int ldc,
    float scale, int do_relu, int c_f32) {
  __shared__ bf16_t As[2][128 * 40];   // [m][k], stride 40
  __shared__ bf16_t Bs[2][256 * 40];   // [n][k], stride 40

  const int tid  = threadIdx.x;
  const int lane = tid & 31;
  const int wave = tid >> 5;
  const int wm   = wave >> 2;               // 0..1
  const int wn   = wave & 3;                // 0..3
  const int m0   = blockIdx.y * 128;
  const int n0   = blockIdx.x * 256;

  // staging geometry: thread owns row tid/2, 16 elems at col (tid&1)*16
  const int srow = tid >> 1;
  const int scol = (tid & 1) * 16;

  v8f acc[4][4];
#pragma unroll
  for (int mf = 0; mf < 4; ++mf)
#pragma unroll
    for (int nf = 0; nf < 4; ++nf)
      acc[mf][nf] = (v8f){0.f, 0.f, 0.f, 0.f, 0.f, 0.f, 0.f, 0.f};

  auto stage = [&](int t, int buf) {
    const bf16_t* ga  = A  + (size_t)(m0 + srow) * lda + t * 32 + scol;
    const bf16_t* gb0 = Bt + (size_t)(n0 + srow) * ldb + t * 32 + scol;
    const bf16_t* gb1 = Bt + (size_t)(n0 + srow + 128) * ldb + t * 32 + scol;
    async_b128(&As[buf][srow * 40 + scol],           ga);
    async_b128(&As[buf][srow * 40 + scol + 8],       ga + 8);
    async_b128(&Bs[buf][srow * 40 + scol],           gb0);
    async_b128(&Bs[buf][srow * 40 + scol + 8],       gb0 + 8);
    async_b128(&Bs[buf][(srow + 128) * 40 + scol],     gb1);
    async_b128(&Bs[buf][(srow + 128) * 40 + scol + 8], gb1 + 8);
  };

  const int T = K >> 5;
  stage(0, 0);
  for (int t = 0; t < T; ++t) {
    const int cur = t & 1;
    if (t + 1 < T) {
      stage(t + 1, cur ^ 1);   // overlap next tile's HBM->LDS with this WMMA
      wait_async_le6();        // tile t's 6 copies landed (in-order per wave)
    } else {
      wait_async_0();
    }
    __syncthreads();

    v16bf bfg[4];
#pragma unroll
    for (int nf = 0; nf < 4; ++nf)
      bfg[nf] = frag_b(&Bs[cur][(wn * 64 + nf * 16) * 40], 40, lane);
#pragma unroll
    for (int mf = 0; mf < 4; ++mf) {
      v16bf af = frag_a(&As[cur][(wm * 64 + mf * 16) * 40], 40, lane);
#pragma unroll
      for (int nf = 0; nf < 4; ++nf)
        acc[mf][nf] = wmma_bf16(af, bfg[nf], acc[mf][nf]);
    }
    __syncthreads();           // readers done before buffer is overwritten
  }

  // ---- epilogue ----
  const int hi = lane >> 4, nC = lane & 15;
#pragma unroll
  for (int mf = 0; mf < 4; ++mf)
#pragma unroll
    for (int nf = 0; nf < 4; ++nf)
#pragma unroll
      for (int r = 0; r < 8; ++r) {
        int row = m0 + wm * 64 + mf * 16 + r + 8 * hi;
        int col = n0 + wn * 64 + nf * 16 + nC;
        float v = acc[mf][nf][r] * scale;
        if (do_relu) v = fmaxf(v, 0.f);
        if (bias) v += bias[col];
        if (c_f32)
          reinterpret_cast<float*>(C)[(size_t)row * ldc + col] = v;
        else
          reinterpret_cast<bf16_t*>(C)[(size_t)row * ldc + col] = (bf16_t)v;
      }
}

// ---------------------------------------------------------------------------
// Flash attention: block = 128 queries of one (b, h); 8 waves x 16 rows.
// Q pre-scaled by 1/8. K tile staged async->LDS; V transposed via registers.
// ---------------------------------------------------------------------------
__global__ __launch_bounds__(256) void mla_attn_kernel(
    const bf16_t* __restrict__ Q, const bf16_t* __restrict__ Km,
    const bf16_t* __restrict__ Vm, bf16_t* __restrict__ O) {
  __shared__ bf16_t Ks[128 * 64];        // [key][dim] == B^T layout for QK^T
  __shared__ bf16_t Vt[64 * 128];        // [dim][key] == B^T layout for P@V
  __shared__ bf16_t Ps[8 * 16 * 128];    // per-wave P tile [row][key]

  const int tid  = threadIdx.x;
  const int lane = tid & 31;
  const int wave = tid >> 5;
  const int hi   = lane >> 4, nC = lane & 15;
  const int q0   = blockIdx.x * 128;
  const int b    = blockIdx.y >> 4;
  const int h    = blockIdx.y & 15;
  const size_t rowbase = (size_t)b * SEQ;
  const int hcol = h * HEAD_DIM;

  // Q fragments in registers for the whole block
  v16bf qa0, qa1;
  {
    const bf16_t* qrow = Q + (rowbase + q0 + wave * 16) * DIM + hcol;
    qa0 = frag_a(qrow, DIM, lane);
    qa1 = frag_a(qrow + 32, DIM, lane);
  }

  float rmax[8], rsum[8];
  v8f o[4];
#pragma unroll
  for (int r = 0; r < 8; ++r) { rmax[r] = -3.0e38f; rsum[r] = 0.f; }
#pragma unroll
  for (int nf = 0; nf < 4; ++nf)
    o[nf] = (v8f){0.f, 0.f, 0.f, 0.f, 0.f, 0.f, 0.f, 0.f};

  bf16_t* myP = &Ps[wave * 16 * 128];
  const int key = tid >> 1;              // staged key row per thread
  const int dmb = (tid & 1) * 32;        // dim chunk base

  for (int kt = 0; kt < SEQ; kt += 128) {
    const size_t gbase = (rowbase + kt + key) * DIM + hcol;
    // K tile: async global->LDS b128 copies
#pragma unroll
    for (int s = 0; s < 4; ++s)
      async_b128(&Ks[key * 64 + dmb + s * 8], Km + gbase + dmb + s * 8);
    // V tile: load + transpose through registers
#pragma unroll
    for (int s = 0; s < 4; ++s) {
      int dm = dmb + s * 8;
      v8bf vv = *reinterpret_cast<const v8bf*>(Vm + gbase + dm);
#pragma unroll
      for (int j = 0; j < 8; ++j) Vt[(dm + j) * 128 + key] = vv[j];
    }
    if (kt + 128 < SEQ) {
      __builtin_prefetch(Km + gbase + 128 * DIM, 0, 3);
      __builtin_prefetch(Vm + gbase + 128 * DIM, 0, 3);
    }
    wait_async_0();
    __syncthreads();

    // ---- S = Q K^T : 16x128 per wave ----
    v8f s[8];
#pragma unroll
    for (int j = 0; j < 8; ++j) {
      v16bf kb0 = frag_b(&Ks[(j * 16) * 64], 64, lane);
      v16bf kb1 = frag_b(&Ks[(j * 16) * 64 + 32], 64, lane);
      v8f t = (v8f){0.f, 0.f, 0.f, 0.f, 0.f, 0.f, 0.f, 0.f};
      t = wmma_bf16(qa0, kb0, t);
      t = wmma_bf16(qa1, kb1, t);
      s[j] = t;
    }

    // ---- online softmax (rows live in 16-lane halves) ----
    float nmax[8], corr[8], psum[8];
#pragma unroll
    for (int r = 0; r < 8; ++r) {
      float m = s[0][r];
#pragma unroll
      for (int j = 1; j < 8; ++j) m = fmaxf(m, s[j][r]);
#pragma unroll
      for (int xm = 1; xm < 16; xm <<= 1) m = fmaxf(m, __shfl_xor(m, xm, 32));
      float nm = fmaxf(rmax[r], m);
      corr[r] = __expf(rmax[r] - nm);
      rmax[r] = nm;
      nmax[r] = nm;
      psum[r] = 0.f;
    }
#pragma unroll
    for (int j = 0; j < 8; ++j)
#pragma unroll
      for (int r = 0; r < 8; ++r) {
        float p = __expf(s[j][r] - nmax[r]);
        psum[r] += p;
        myP[(r + 8 * hi) * 128 + j * 16 + nC] = (bf16_t)p;
      }
#pragma unroll
    for (int r = 0; r < 8; ++r) {
      float ps = psum[r];
#pragma unroll
      for (int xm = 1; xm < 16; xm <<= 1) ps += __shfl_xor(ps, xm, 32);
      rsum[r] = rsum[r] * corr[r] + ps;
#pragma unroll
      for (int nf = 0; nf < 4; ++nf) o[nf][r] = o[nf][r] * corr[r];
    }

    // ---- O += P @ V (same-wave LDS RAW is in-order) ----
#pragma unroll
    for (int ko = 0; ko < 4; ++ko) {
      v16bf pa = frag_a(myP + ko * 32, 128, lane);
#pragma unroll
      for (int nf = 0; nf < 4; ++nf) {
        v16bf vb = frag_b(&Vt[(nf * 16) * 128 + ko * 32], 128, lane);
        o[nf] = wmma_bf16(pa, vb, o[nf]);
      }
    }
    __syncthreads();
  }

  // ---- normalize + store to (b, q, h*64+dim) ----
#pragma unroll
  for (int nf = 0; nf < 4; ++nf)
#pragma unroll
    for (int r = 0; r < 8; ++r) {
      int row = q0 + wave * 16 + r + 8 * hi;
      int col = hcol + nf * 16 + nC;
      O[(rowbase + row) * DIM + col] = (bf16_t)(o[nf][r] / rsum[r]);
    }
}

// ---------------------------------------------------------------------------
// Host-side orchestration
// ---------------------------------------------------------------------------
extern "C" void kernel_launch(void* const* d_in, const int* in_sizes, int n_in,
                              void* d_out, int out_size, void* d_ws, size_t ws_size,
                              hipStream_t stream) {
  const float* x     = (const float*)d_in[0];
  const float* w_q   = (const float*)d_in[1];
  const float* w_kv  = (const float*)d_in[2];
  const float* w_lk1 = (const float*)d_in[3];
  const float* w_lk2 = (const float*)d_in[4];
  const float* w_lv1 = (const float*)d_in[5];
  const float* w_lv2 = (const float*)d_in[6];
  const float* w_out = (const float*)d_in[7];
  const float* b_out = (const float*)d_in[8];

  char* ws = (char*)d_ws;
  size_t off = 0;
  auto alloc = [&](size_t elems) -> bf16_t* {
    bf16_t* p = (bf16_t*)(ws + off);
    off = (off + elems * sizeof(bf16_t) + 255) & ~(size_t)255;
    return p;
  };

  bf16_t* xbf    = alloc((size_t)ROWS * DIM);
  bf16_t* wqT    = alloc((size_t)DIM * DIM);        // [N=1024][K=1024]
  bf16_t* wkvT   = alloc((size_t)2 * DIM * DIM);    // [N=2048][K=1024]
  bf16_t* lk1T   = alloc((size_t)LATENT * DIM);     // [512][1024]
  bf16_t* lk2T   = alloc((size_t)DIM * LATENT);     // [1024][512]
  bf16_t* lv1T   = alloc((size_t)LATENT * DIM);
  bf16_t* lv2T   = alloc((size_t)DIM * LATENT);
  bf16_t* woutT  = alloc((size_t)DIM * DIM);
  bf16_t* qbf    = alloc((size_t)ROWS * DIM);
  bf16_t* kbuf   = alloc((size_t)ROWS * DIM);       // k0, then final k
  bf16_t* vbuf   = alloc((size_t)ROWS * DIM);       // v0, then final v
  bf16_t* lbuf   = alloc((size_t)ROWS * LATENT);    // latent scratch
  bf16_t* attn   = alloc((size_t)ROWS * DIM);

  {
    int n4 = ROWS * DIM / 4;
    f32_to_bf16_kernel<<<(n4 + 255) / 256, 256, 0, stream>>>(x, xbf, n4);
  }
  auto transp = [&](const float* src, bf16_t* dst, int K, int N) {
    transpose_f32_to_bf16_kernel<<<dim3(N / 32, K / 32), 256, 0, stream>>>(
        src, dst, K, N);
  };
  transp(w_q,   wqT,   DIM, DIM);
  transp(w_kv,  wkvT,  DIM, 2 * DIM);
  transp(w_lk1, lk1T,  DIM, LATENT);
  transp(w_lk2, lk2T,  LATENT, DIM);
  transp(w_lv1, lv1T,  DIM, LATENT);
  transp(w_lv2, lv2T,  LATENT, DIM);
  transp(w_out, woutT, DIM, DIM);

  auto gemm = [&](const bf16_t* A, const bf16_t* Bt, void* C, const float* bias,
                  int M, int N, int K, int lda, int ldb, int ldc,
                  float scale, int relu, int cf32) {
    dim3 g(N / 256, M / 128);
    gemm_bf16_kernel<<<g, 256, 0, stream>>>(A, Bt, C, bias, M, N, K,
                                            lda, ldb, ldc, scale, relu, cf32);
  };

  const float qscale = 0.125f;  // HEAD_DIM^-0.5 folded into q projection

  gemm(xbf, wqT, qbf, nullptr, ROWS, DIM, DIM, DIM, DIM, DIM, qscale, 0, 0);
  gemm(xbf, wkvT,             kbuf, nullptr, ROWS, DIM, DIM, DIM, DIM, DIM, 1.f, 0, 0);
  gemm(xbf, wkvT + (size_t)DIM * DIM, vbuf, nullptr, ROWS, DIM, DIM, DIM, DIM, DIM, 1.f, 0, 0);
  gemm(kbuf, lk1T, lbuf, nullptr, ROWS, LATENT, DIM, DIM, DIM, LATENT, 1.f, 1, 0);
  gemm(lbuf, lk2T, kbuf, nullptr, ROWS, DIM, LATENT, LATENT, LATENT, DIM, 1.f, 0, 0);
  gemm(vbuf, lv1T, lbuf, nullptr, ROWS, LATENT, DIM, DIM, DIM, LATENT, 1.f, 1, 0);
  gemm(lbuf, lv2T, vbuf, nullptr, ROWS, DIM, LATENT, LATENT, LATENT, DIM, 1.f, 0, 0);

  mla_attn_kernel<<<dim3(SEQ / 128, BATCH * HEADS), 256, 0, stream>>>(
      qbf, kbuf, vbuf, attn);

  gemm(attn, woutT, d_out, b_out, ROWS, DIM, DIM, DIM, DIM, DIM, 1.f, 0, 1);

  (void)in_sizes; (void)n_in; (void)out_size; (void)ws_size;
}
